// LocalWindowMultiHeadAttentionFast_11338713661445
// MI455X (gfx1250) — compile-verified, matrix operand order checked by reading, not run on hardware
//
#include <hip/hip_runtime.h>
#include <hip/hip_bf16.h>

typedef __bf16  v16bf __attribute__((ext_vector_type(16)));
typedef float   v8f   __attribute__((ext_vector_type(8)));

#define BB 2
#define HH 112
#define WW 112
#define CC 128
#define NH 4
#define HD 32

__device__ __forceinline__ int reflect_idx(int i, int n) {
    if (i < 0) i = -i;
    if (i >= n) i = 2 * n - 2 - i;
    return i;
}

__device__ __forceinline__ v8f wmma_bf16(v16bf a, v16bf b, v8f c) {
    return __builtin_amdgcn_wmma_f32_16x16x32_bf16(false, a, false, b, (short)0, c, false, false);
}

// ---------------------------------------------------------------------------
// Kernel 0: convert + transpose weights to bf16 [N][K] layout so WMMA B
// operands are contiguous 32B per lane.
// ---------------------------------------------------------------------------
__global__ void cvt_weights(const float* __restrict__ wqkv,  // [128][384]
                            const float* __restrict__ wout,  // [128][128]
                            __bf16* __restrict__ wqkvT,      // [384][128]
                            __bf16* __restrict__ woutT) {    // [128][128]
    int idx = blockIdx.x * 256 + threadIdx.x;
    if (idx < 384 * 128) {
        int n = idx / 128, k = idx % 128;
        wqkvT[idx] = (__bf16)wqkv[k * 384 + n];
    } else {
        int i = idx - 384 * 128;
        if (i < 128 * 128) {
            int n = i / 128, k = i % 128;
            woutT[i] = (__bf16)wout[k * 128 + n];
        }
    }
}

// ---------------------------------------------------------------------------
// Kernel 1: QKV projection GEMM. M=25088 pixels, N=384 (3 groups of 128),
// K=128. Block = 8 waves; wave computes a 16x128 strip. fp32 x -> bf16 A
// on the fly; acc fp32; fused bias; bf16 store into split q/k/v buffers.
// ---------------------------------------------------------------------------
__global__ void qkv_gemm(const float* __restrict__ x,
                         const __bf16* __restrict__ wT,     // [384][128]
                         const float* __restrict__ bias,    // [384]
                         __bf16* __restrict__ qb,
                         __bf16* __restrict__ kb,
                         __bf16* __restrict__ vb) {
    const int lane = threadIdx.x & 31;
    const int wv   = threadIdx.x >> 5;
    const int ln   = lane & 15;
    const int half = lane >> 4;
    const int g    = blockIdx.y;                 // 0=q 1=k 2=v
    const int m0   = blockIdx.x * 128 + wv * 16; // base pixel row for wave
    const int mrow = m0 + ln;                    // A-operand row for this lane

    v8f acc[8];
#pragma unroll
    for (int nt = 0; nt < 8; ++nt) acc[nt] = (v8f){0.f,0.f,0.f,0.f,0.f,0.f,0.f,0.f};

#pragma unroll
    for (int kk = 0; kk < 4; ++kk) {
        const int k0 = kk * 32 + half * 16;
        const float4* ap = (const float4*)(x + (size_t)mrow * CC + k0);
        float4 a0 = ap[0], a1 = ap[1], a2 = ap[2], a3 = ap[3];
        float af[16] = {a0.x,a0.y,a0.z,a0.w, a1.x,a1.y,a1.z,a1.w,
                        a2.x,a2.y,a2.z,a2.w, a3.x,a3.y,a3.z,a3.w};
        v16bf av;
#pragma unroll
        for (int i = 0; i < 16; ++i) av[i] = (__bf16)af[i];
#pragma unroll
        for (int nt = 0; nt < 8; ++nt) {
            const v16bf bv = *(const v16bf*)(wT + (size_t)(g * 128 + nt * 16 + ln) * CC + k0);
            acc[nt] = wmma_bf16(av, bv, acc[nt]);
        }
    }

    __bf16* dst = (g == 0) ? qb : ((g == 1) ? kb : vb);
#pragma unroll
    for (int nt = 0; nt < 8; ++nt) {
        const int c = nt * 16 + ln;
        const float bz = bias[g * 128 + c];
#pragma unroll
        for (int r = 0; r < 8; ++r) {
            const int row = m0 + r + half * 8;
            dst[(size_t)row * CC + c] = (__bf16)(acc[nt][r] + bz);
        }
    }
}

// ---------------------------------------------------------------------------
// Kernel 2: union-window attention. Block = 4 waves (one head each) for a
// 16-pixel row segment. Union = 5 rows x 20 cols = 100 positions (pad 128).
//   scores[16x112] = Q[16x32] @ Kt        (7 WMMAs)
//   mask + row softmax (shfl_xor width-16 matches C/D half-wave layout)
//   P -> LDS (A-layout), V^T staged in LDS
//   out[16x32]   = P[16x128] @ V          (8 WMMAs)
// ---------------------------------------------------------------------------
__global__ void __launch_bounds__(128)
attn_kernel(const __bf16* __restrict__ qb,
            const __bf16* __restrict__ kb,
            const __bf16* __restrict__ vb,
            __bf16* __restrict__ ob) {
    __shared__ __bf16 smem[4 * 2048 + 4 * 4096];   // P: 4x16x128, VT: 4x32x128
    const int lane = threadIdx.x & 31;
    const int wv   = threadIdx.x >> 5;             // head
    const int ln   = lane & 15;
    const int half = lane >> 4;
    const int x0   = blockIdx.x * 16;
    const int y    = blockIdx.y;
    const int b    = blockIdx.z;

    __bf16* P  = smem + wv * 2048;
    __bf16* VT = smem + 4 * 2048 + wv * 4096;

    // ---- Q A-operand: lane holds pixel x0+ln, dims half*16..+15 (32B) ----
    const size_t qoff = ((size_t)(b * HH + y) * WW + x0 + ln) * CC + wv * HD + half * 16;
    const v16bf qa = *(const v16bf*)(qb + qoff);

    // ---- scores: 7 WMMAs over the 112-position union ----
    v8f s[7];
#pragma unroll
    for (int t = 0; t < 7; ++t) {
        const int j = t * 16 + ln;
        int py = y, px = x0;
        if (j < 100) {
            const int r = j / 20, c = j % 20;
            py = reflect_idx(y - 2 + r, HH);
            px = reflect_idx(x0 - 2 + c, WW);
        }
        const v16bf kv = *(const v16bf*)(kb + ((size_t)(b * HH + py) * WW + px) * CC + wv * HD + half * 16);
        v8f z = (v8f){0.f,0.f,0.f,0.f,0.f,0.f,0.f,0.f};
        s[t] = wmma_bf16(qa, kv, z);
    }

    // ---- mask + softmax per output row m = r + half*8 ----
    const float scale = 0.17677669529663687f;  // 1/sqrt(32)
    int   ccj[7];
    bool  jvv[7];
#pragma unroll
    for (int t = 0; t < 7; ++t) { const int j = t * 16 + ln; ccj[t] = j % 20; jvv[t] = (j < 100); }

#pragma unroll
    for (int r = 0; r < 8; ++r) {
        const int m = r + half * 8;
        float vals[7];
        float mx = -1e30f;
#pragma unroll
        for (int t = 0; t < 7; ++t) {
            const bool ok = jvv[t] && (ccj[t] >= m) && (ccj[t] <= m + 4);
            vals[t] = ok ? s[t][r] * scale : -1e30f;
            mx = fmaxf(mx, vals[t]);
        }
        mx = fmaxf(mx, __shfl_xor(mx, 8, 16));
        mx = fmaxf(mx, __shfl_xor(mx, 4, 16));
        mx = fmaxf(mx, __shfl_xor(mx, 2, 16));
        mx = fmaxf(mx, __shfl_xor(mx, 1, 16));
        float ev[7];
        float sum = 0.f;
#pragma unroll
        for (int t = 0; t < 7; ++t) {
            const float e = (vals[t] > -1e29f) ? __expf(vals[t] - mx) : 0.f;
            ev[t] = e; sum += e;
        }
        sum += __shfl_xor(sum, 8, 16);
        sum += __shfl_xor(sum, 4, 16);
        sum += __shfl_xor(sum, 2, 16);
        sum += __shfl_xor(sum, 1, 16);
        const float inv = 1.f / sum;
#pragma unroll
        for (int t = 0; t < 7; ++t)
            P[m * 128 + t * 16 + ln] = (__bf16)(ev[t] * inv);
        P[m * 128 + 112 + ln] = (__bf16)0.f;   // zero K-pad columns
    }

    // ---- stage V^T in LDS: VT[d][j]; global read is 64B coalesced per j ----
    for (int j = 0; j < 112; ++j) {
        int py = y, px = x0;
        if (j < 100) {
            const int r = j / 20, c = j % 20;
            py = reflect_idx(y - 2 + r, HH);
            px = reflect_idx(x0 - 2 + c, WW);
        }
        VT[lane * 128 + j] = vb[((size_t)(b * HH + py) * WW + px) * CC + wv * HD + lane];
    }
#pragma unroll
    for (int j = 112; j < 128; ++j) VT[lane * 128 + j] = (__bf16)0.f;

    __syncthreads();

    // ---- out = P @ V : K=128 in 4 chunks, N=32 in 2 tiles ----
    v8f o[2];
    o[0] = (v8f){0.f,0.f,0.f,0.f,0.f,0.f,0.f,0.f};
    o[1] = (v8f){0.f,0.f,0.f,0.f,0.f,0.f,0.f,0.f};
#pragma unroll
    for (int kc = 0; kc < 4; ++kc) {
        const v16bf pa = *(const v16bf*)(P + ln * 128 + kc * 32 + half * 16);
#pragma unroll
        for (int nt = 0; nt < 2; ++nt) {
            const v16bf vv = *(const v16bf*)(VT + (nt * 16 + ln) * 128 + kc * 32 + half * 16);
            o[nt] = wmma_bf16(pa, vv, o[nt]);
        }
    }

    __bf16* op = ob + (size_t)(b * HH + y) * WW * CC + wv * HD;
#pragma unroll
    for (int nt = 0; nt < 2; ++nt)
#pragma unroll
        for (int r = 0; r < 8; ++r)
            op[(size_t)(x0 + r + half * 8) * CC + nt * 16 + ln] = (__bf16)o[nt][r];
}

// ---------------------------------------------------------------------------
// Kernel 3: output projection GEMM [25088x128] @ [128x128] + bias, fp32 out.
// ---------------------------------------------------------------------------
__global__ void out_gemm(const __bf16* __restrict__ a,
                         const __bf16* __restrict__ wT,   // [128][128]
                         const float* __restrict__ bias,  // [128]
                         float* __restrict__ out) {
    const int lane = threadIdx.x & 31;
    const int wv   = threadIdx.x >> 5;
    const int ln   = lane & 15;
    const int half = lane >> 4;
    const int m0   = blockIdx.x * 128 + wv * 16;
    const int mrow = m0 + ln;

    v8f acc[8];
#pragma unroll
    for (int nt = 0; nt < 8; ++nt) acc[nt] = (v8f){0.f,0.f,0.f,0.f,0.f,0.f,0.f,0.f};

#pragma unroll
    for (int kk = 0; kk < 4; ++kk) {
        const int k0 = kk * 32 + half * 16;
        const v16bf av = *(const v16bf*)(a + (size_t)mrow * CC + k0);
#pragma unroll
        for (int nt = 0; nt < 8; ++nt) {
            const v16bf bv = *(const v16bf*)(wT + (size_t)(nt * 16 + ln) * CC + k0);
            acc[nt] = wmma_bf16(av, bv, acc[nt]);
        }
    }

#pragma unroll
    for (int nt = 0; nt < 8; ++nt) {
        const int c = nt * 16 + ln;
        const float bz = bias[c];
#pragma unroll
        for (int r = 0; r < 8; ++r)
            out[(size_t)(m0 + r + half * 8) * CC + c] = acc[nt][r] + bz;
    }
}

// ---------------------------------------------------------------------------
extern "C" void kernel_launch(void* const* d_in, const int* in_sizes, int n_in,
                              void* d_out, int out_size, void* d_ws, size_t ws_size,
                              hipStream_t stream) {
    const float* x      = (const float*)d_in[0];
    const float* w_qkv  = (const float*)d_in[1];
    const float* b_qkv  = (const float*)d_in[2];
    const float* w_out  = (const float*)d_in[3];
    const float* b_out  = (const float*)d_in[4];

    char* ws = (char*)d_ws;
    const size_t PIX   = (size_t)BB * HH * WW;          // 25088
    const size_t QKVSZ = PIX * CC * sizeof(__bf16);     // 6,422,528 B

    __bf16* wqkvT = (__bf16*)(ws);                       // 98,304 B
    __bf16* woutT = (__bf16*)(ws + 98304);               // 32,768 B
    __bf16* qb    = (__bf16*)(ws + 131072);
    __bf16* kb    = (__bf16*)(ws + 131072 + QKVSZ);
    __bf16* vb    = (__bf16*)(ws + 131072 + 2 * QKVSZ);
    __bf16* ab    = (__bf16*)(ws + 131072 + 3 * QKVSZ);

    cvt_weights<<<256, 256, 0, stream>>>(w_qkv, w_out, wqkvT, woutT);
    qkv_gemm<<<dim3(196, 3), 256, 0, stream>>>(x, wqkvT, b_qkv, qb, kb, vb);
    attn_kernel<<<dim3(WW / 16, HH, BB), 128, 0, stream>>>(qb, kb, vb, ab);
    out_gemm<<<196, 256, 0, stream>>>(ab, woutT, b_out, (float*)d_out);
}
